// discConv_50465865728280
// MI455X (gfx1250) — matrix operand order repaired and checked by reading, not compiled, not run on hardware
//
#include <hip/hip_runtime.h>

// ---------- problem constants ----------
#define FI   256
#define FO   512
#define Bn   8
#define Hs   256
#define Ws_  256
#define HP   257          // FIR output spatial
#define OH   128
#define OW   128
#define W_SCALE (1.0f/48.0f)   // (FI*3*3)^-0.5 = 1/sqrt(2304)

typedef __bf16    bf16x16 __attribute__((ext_vector_type(16)));
typedef float     f32x8   __attribute__((ext_vector_type(8)));
typedef unsigned  u32x4   __attribute__((ext_vector_type(4)));

union FragBF { bf16x16 v; u32x4 q[2]; };

__device__ __forceinline__ unsigned short f2bf(float f) {
    unsigned u = __builtin_bit_cast(unsigned, f);
    u += 0x7FFFu + ((u >> 16) & 1u);          // round-to-nearest-even
    return (unsigned short)(u >> 16);
}

// ============================================================
// Stage 1: depthwise 4x4 FIR, NCHW fp32 -> NHWC bf16 (padded to 257x257)
// ============================================================
__global__ __launch_bounds__(256) void fir_kernel(
    const float* __restrict__ x, const float* __restrict__ fir,
    unsigned short* __restrict__ y2)
{
    __shared__ float xs[4 * 32 * 69];   // [i=4][f=32][w=69(pad)] = 35.3 KB
    const int w0 = blockIdx.x * 64;
    const int h  = blockIdx.y;          // 0..256
    const int b  = blockIdx.z;
    const int t  = threadIdx.x;

    float fc[16];
#pragma unroll
    for (int n = 0; n < 16; ++n) fc[n] = fir[n];

    for (int fi0 = 0; fi0 < FI; fi0 += 32) {
        __syncthreads();
        for (int idx = t; idx < 128 * 67; idx += 256) {
            int row = idx / 67;
            int j   = idx - row * 67;
            int f   = row >> 2;
            int i   = row & 3;
            int hh  = h - 2 + i;
            int cc  = w0 - 2 + j;
            float v = 0.0f;
            if ((unsigned)hh < (unsigned)Hs && (unsigned)cc < (unsigned)Ws_)
                v = x[(((size_t)b * FI + (fi0 + f)) * Hs + hh) * Ws_ + cc];
            xs[(i * 32 + f) * 69 + j] = v;
        }
        __syncthreads();

        const int fp = t & 15;
        const int f  = fp * 2;
        const int wb = t >> 4;
#pragma unroll
        for (int k = 0; k < 4; ++k) {
            int w  = wb + 16 * k;
            int wg = w0 + w;
            if (wg < HP) {
                float a0 = 0.0f, a1 = 0.0f;
#pragma unroll
                for (int i = 0; i < 4; ++i)
#pragma unroll
                    for (int j = 0; j < 4; ++j) {
                        float c = fc[i * 4 + j];
                        a0 += c * xs[(i * 32 + f    ) * 69 + (w + j)];
                        a1 += c * xs[(i * 32 + f + 1) * 69 + (w + j)];
                    }
                unsigned pk = (unsigned)f2bf(a0) | ((unsigned)f2bf(a1) << 16);
                size_t base = (((size_t)b * HP + h) * HP + wg) * FI + fi0 + f;
                *(unsigned*)(y2 + base) = pk;
            }
        }
    }
}

// ============================================================
// Stage 1.5: weights -> wt2[kh*3+kw][fo][fi] bf16, pre-scaled
// ============================================================
__global__ __launch_bounds__(256) void wprep_kernel(
    const float* __restrict__ w, unsigned short* __restrict__ wt2)
{
    int idx = blockIdx.x * 256 + threadIdx.x;   // = (t9*512 + fo)*256 + fi
    if (idx >= 9 * FO * FI) return;
    int fi = idx & 255;
    int fo = (idx >> 8) & 511;
    int t9 = idx >> 17;
    float v = w[((size_t)(fo * FI + fi)) * 9 + t9] * W_SCALE;
    wt2[idx] = f2bf(v);
}

// ============================================================
// Stage 2: implicit GEMM, bf16 WMMA, double-buffered LDS with
// GLOBAL_LOAD_ASYNC_TO_LDS_B128 prefetch (ASYNCcnt-tracked DMA).
// Per-thread K-walk uses incremental pointers (piecewise-constant deltas),
// so the steady-state loop has no address recomputation and no branches.
// ============================================================
#define LDSROW 40           // ushorts per LDS row (32 data + 8 pad -> 80B stride)
#define TILEU  (128*LDSROW) // ushorts per tile buffer
#define NK     72           // 9 taps * 8 fi-chunks

__global__ __launch_bounds__(256) void gemm_kernel(
    const unsigned short* __restrict__ y2,
    const unsigned short* __restrict__ wt2,
    const float* __restrict__ bias,
    float* __restrict__ out)
{
    __shared__ unsigned short Wsh[2][TILEU];  // A tile: [fo][fi]
    __shared__ unsigned short Xsh[2][TILEU];  // B tile: [ow][fi]

    const int fo0  = blockIdx.x * 128;
    const int oh   = blockIdx.y;
    const int b    = blockIdx.z;
    const int t    = threadIdx.x;
    const int lane = t & 31;
    const int lo   = lane & 15;
    const int hi   = lane >> 4;
    const int wv   = t >> 5;
    const int wm   = wv & 1;   // fo half (64)
    const int wn   = wv >> 1;  // ow quarter (32)

    // tile-fill role: threads 0-127 fill W tile, 128-255 fill X tile
    const bool isW = (t < 128);
    const int  r   = isW ? t : (t - 128);
    const unsigned ldsdst0 = isW
        ? (unsigned)(unsigned long long)(void*)&Wsh[0][r * LDSROW]
        : (unsigned)(unsigned long long)(void*)&Xsh[0][r * LDSROW];

    // Incremental global source pointer for this thread's 64B slice.
    // Within a tap: +32 ushorts. Tap boundary: role-dependent constant;
    // kh boundary (after ks==23/47) is a row jump for X threads.
    const unsigned short* src;
    int dT9, dRow;
    if (isW) {
        src  = wt2 + (size_t)(fo0 + r) * FI;
        dT9  = FO * FI - 224;
        dRow = dT9;
    } else {
        src  = y2 + ((size_t)b * HP + 2 * oh) * HP * FI + (size_t)(2 * r) * FI;
        dT9  = FI - 224;
        dRow = (HP - 2) * FI - 224;
    }

    auto prefetch = [&](int buf) {   // 64B per thread = 4x B128, shared addr pair
        const unsigned dst = ldsdst0 + (unsigned)buf * (TILEU * 2);
        asm volatile("global_load_async_to_lds_b128 %0, %1, off"
                     :: "v"(dst), "v"(src) : "memory");
        asm volatile("global_load_async_to_lds_b128 %0, %1, off offset:16"
                     :: "v"(dst), "v"(src) : "memory");
        asm volatile("global_load_async_to_lds_b128 %0, %1, off offset:32"
                     :: "v"(dst), "v"(src) : "memory");
        asm volatile("global_load_async_to_lds_b128 %0, %1, off offset:48"
                     :: "v"(dst), "v"(src) : "memory");
    };
    auto advance = [&](int k) {      // move src from K-step k to k+1
        src += ((k & 7) != 7) ? 32 : ((k == 23 || k == 47) ? dRow : dT9);
    };

    f32x8 acc[4][2];
#pragma unroll
    for (int i = 0; i < 4; ++i)
#pragma unroll
        for (int j = 0; j < 2; ++j) acc[i][j] = {};

    auto kstep = [&](int ks, int buf) {
        // my async writes landed in LDS, then sync the workgroup
        asm volatile("s_wait_asynccnt 0x0" ::: "memory");
        __syncthreads();
        if (ks + 1 < NK) {           // overlaps with WMMA below
            prefetch(buf ^ 1);
            advance(ks + 1);
        }
        const unsigned short* Wb = Wsh[buf];
        const unsigned short* Xb = Xsh[buf];

        // A fragments (16x32 bf16): lanes 0-15 row M=l, K 0-7 & 16-23;
        // lanes 16-31 same rows, K 8-15 & 24-31.
        FragBF af[4];
#pragma unroll
        for (int fm = 0; fm < 4; ++fm) {
            const unsigned short* p = Wb + (wm * 64 + fm * 16 + lo) * LDSROW + hi * 8;
            af[fm].q[0] = *(const u32x4*)p;
            af[fm].q[1] = *(const u32x4*)(p + 16);
        }
        // B fragments (32x16): lane = col N; lanes 0-15 K 0-7 & 16-23, lanes 16-31 K 8-15 & 24-31.
        FragBF bfr[2];
#pragma unroll
        for (int fn = 0; fn < 2; ++fn) {
            const unsigned short* p = Xb + (wn * 32 + fn * 16 + lo) * LDSROW + hi * 8;
            bfr[fn].q[0] = *(const u32x4*)p;
            bfr[fn].q[1] = *(const u32x4*)(p + 16);
        }
        // keep all 12 ds_loads ahead of the 8 WMMAs -> single dscnt wait
        __builtin_amdgcn_sched_barrier(0);
#pragma unroll
        for (int fm = 0; fm < 4; ++fm)
#pragma unroll
            for (int fn = 0; fn < 2; ++fn)
                acc[fm][fn] = __builtin_amdgcn_wmma_f32_16x16x32_bf16(
                    false, af[fm].v, false, bfr[fn].v,
                    (short)0, acc[fm][fn], false, false);
    };

    prefetch(0);
    advance(0);
    for (int ks = 0; ks < NK; ks += 2) {
        kstep(ks,     0);
        kstep(ks + 1, 1);
    }

    // Epilogue: bias + LeakyReLU(0.2). D layout: lane: N=lo, VGPR v: M=v+8*hi.
#pragma unroll
    for (int fm = 0; fm < 4; ++fm)
#pragma unroll
        for (int fn = 0; fn < 2; ++fn) {
            int ow = wn * 32 + fn * 16 + lo;
#pragma unroll
            for (int v = 0; v < 8; ++v) {
                int fo = fo0 + wm * 64 + fm * 16 + v + 8 * hi;
                float val = acc[fm][fn][v] + bias[fo];
                val = val >= 0.0f ? val : 0.2f * val;
                out[(((size_t)b * FO + fo) * OH + oh) * OW + ow] = val;
            }
        }
}

// ============================================================
extern "C" void kernel_launch(void* const* d_in, const int* in_sizes, int n_in,
                              void* d_out, int out_size, void* d_ws, size_t ws_size,
                              hipStream_t stream)
{
    (void)in_sizes; (void)n_in; (void)out_size; (void)ws_size;
    const float* x    = (const float*)d_in[0];
    const float* w    = (const float*)d_in[1];
    const float* bias = (const float*)d_in[2];
    const float* fir  = (const float*)d_in[3];
    float* out = (float*)d_out;

    unsigned short* y2  = (unsigned short*)d_ws;                 // 8*257*257*256 bf16
    unsigned short* wt2 = y2 + (size_t)Bn * HP * HP * FI;        // 9*512*256 bf16

    fir_kernel  <<<dim3(5, HP, Bn), 256, 0, stream>>>(x, fir, y2);
    wprep_kernel<<<dim3((9 * FO * FI) / 256), 256, 0, stream>>>(w, wt2);
    gemm_kernel <<<dim3(FO / 128, OH, Bn), 256, 0, stream>>>(y2, wt2, bias, out);
}
